// OCRAttention_86251533238782
// MI455X (gfx1250) — compile-verified, matrix operand order checked by reading, not run on hardware
//
#include <hip/hip_runtime.h>

// ---------------------------------------------------------------------------
// OCR attention, MI455X (gfx1250, wave32, WMMA).
// bf16 WMMA (f32 accum) for all GEMMs; flash-style online softmax so the
// [B,H,S,S] score tensor (537 MB) is never materialized.
// Round 5: sched_group_barrier directives in the GEMM k-loops pin the
// schedule to "10 VMEM reads, then 4 WMMAs" per k-step, forcing disjoint
// fragment registers and partial s_wait_loadcnt (load/WMMA overlap) that
// source-level pipelining could not achieve against the re-rolling scheduler.
// ---------------------------------------------------------------------------

typedef __attribute__((ext_vector_type(16))) __bf16 v16bf;
typedef __attribute__((ext_vector_type(8)))  float  v8f;

#define S_LEN 2048
#define DM    512
#define NH    8
#define HD    64
#define BATCH 4
#define NTOK  (BATCH * S_LEN)   // 8192

__device__ __forceinline__ v8f wmma_bf16(v16bf a, v16bf b, v8f c) {
  // v_wmma_f32_16x16x32_bf16  D = A(16x32) * B(32x16) + C(16x16 f32)
  return __builtin_amdgcn_wmma_f32_16x16x32_bf16(
      /*neg_a=*/false, a, /*neg_b=*/false, b,
      /*c_mod=*/(short)0, c, /*reuse_a=*/false, /*reuse_b=*/false);
}

// Pin per-k-step schedule: VMEM-read group (mask 0x20) of 10 b128 loads
// ahead of an MFMA/WMMA group (mask 0x8) of 4 matrix ops.
__device__ __forceinline__ void sched_loads_then_wmma() {
#if __has_builtin(__builtin_amdgcn_sched_group_barrier)
  __builtin_amdgcn_sched_group_barrier(0x20, 10, 0);  // 10 VMEM reads
  __builtin_amdgcn_sched_group_barrier(0x08,  4, 0);  // 4 WMMAs
#endif
}

// ---------------------------------------------------------------------------
// f32 -> bf16 elementwise convert
// ---------------------------------------------------------------------------
__global__ void f32_to_bf16_kernel(const float* __restrict__ src,
                                   __bf16* __restrict__ dst, int n) {
  int i = blockIdx.x * blockDim.x + threadIdx.x;
  if (i < n) dst[i] = (__bf16)src[i];
}

// ---------------------------------------------------------------------------
// QKV projection: out = x @ W^T + bias.  One wave per 16x64 output tile:
// 4 f32 accumulators share the same A-frags (x rows); B-frags come straight
// from W rows (W rows ARE B columns since we multiply by W^T).
// z = 0 -> Q (+ocr_bias) stored [b,h,s,d];  z = 1 -> K stored [b,h,s,d];
// z = 2 -> V stored transposed [b,h,d,s] so PV B-frags are contiguous.
// ---------------------------------------------------------------------------
__global__ void qkv_proj_kernel(const __bf16* __restrict__ xb,
                                const __bf16* __restrict__ wqb,
                                const __bf16* __restrict__ wkb,
                                const __bf16* __restrict__ wvb,
                                const float* __restrict__ bq,
                                const float* __restrict__ bk,
                                const float* __restrict__ bv,
                                const float* __restrict__ ocr_bias,
                                __bf16* __restrict__ qhb,
                                __bf16* __restrict__ khb,
                                __bf16* __restrict__ vTb) {
  const int lane = threadIdx.x & 31;
  const int half = lane >> 4;        // 0: lanes 0-15, 1: lanes 16-31
  const int lm   = lane & 15;
  const int colbase = blockIdx.x * 64;
  const int rowbase = blockIdx.y * 16;
  const int z = blockIdx.z;

  const __bf16* W = (z == 0) ? wqb : (z == 1) ? wkb : wvb;
  const __bf16* arow = xb + (size_t)(rowbase + lm) * DM + half * 16;
  const __bf16* brow = W  + (size_t)(colbase + lm) * DM + half * 16;

  v8f acc0 = {}, acc1 = {}, acc2 = {}, acc3 = {};

  #pragma unroll 2
  for (int kk = 0; kk < DM; kk += 32) {
    v16bf a  = *(const v16bf*)(arow + kk);
    v16bf b0 = *(const v16bf*)(brow + (size_t)(0 * 16) * DM + kk);
    v16bf b1 = *(const v16bf*)(brow + (size_t)(1 * 16) * DM + kk);
    v16bf b2 = *(const v16bf*)(brow + (size_t)(2 * 16) * DM + kk);
    v16bf b3 = *(const v16bf*)(brow + (size_t)(3 * 16) * DM + kk);
    acc0 = wmma_bf16(a, b0, acc0);
    acc1 = wmma_bf16(a, b1, acc1);
    acc2 = wmma_bf16(a, b2, acc2);
    acc3 = wmma_bf16(a, b3, acc3);
    sched_loads_then_wmma();
  }

  v8f acc[4] = {acc0, acc1, acc2, acc3};
  #pragma unroll
  for (int nt = 0; nt < 4; ++nt) {
    const int c = colbase + nt * 16 + lm;   // output channel for this lane
    const int h = c >> 6, d = c & 63;
    #pragma unroll
    for (int r = 0; r < 8; ++r) {
      const int m = r + 8 * half;           // C/D layout row
      const int t = rowbase + m;            // global token
      const int b_ = t >> 11;               // /S_LEN
      const int s  = t & (S_LEN - 1);
      float v = acc[nt][r];
      if (z == 0) {
        v += bq[c] + ocr_bias[c];
        qhb[((size_t)(b_ * NH + h) * S_LEN + s) * HD + d] = (__bf16)v;
      } else if (z == 1) {
        v += bk[c];
        khb[((size_t)(b_ * NH + h) * S_LEN + s) * HD + d] = (__bf16)v;
      } else {
        v += bv[c];
        vTb[((size_t)(b_ * NH + h) * HD + d) * S_LEN + s] = (__bf16)v;
      }
    }
  }
}

// ---------------------------------------------------------------------------
// Flash attention: one wave per (b, h, 32-query tile).  Two Q-frag sets and
// two softmax states; K/V fragments are loaded once and reused by both.
// Per 32-key step: 8 score WMMAs, banded bias + online softmax (16-lane shfl
// butterflies for row max/sum), P routed through a 2KB LDS tile (f32 D-layout
// -> bf16 A-layout), then 8 PV WMMAs into two 16x64 f32 accumulators.
// K/V for the NEXT key block are prefetched with full-wave lane coverage.
// ---------------------------------------------------------------------------
__global__ void attn_fwd_kernel(const __bf16* __restrict__ qhb,
                                const __bf16* __restrict__ khb,
                                const __bf16* __restrict__ vTb,
                                __bf16* __restrict__ ctx) {
  __shared__ __align__(32) __bf16 pbuf[32 * 32];   // 32 queries x 32 keys

  const int lane = threadIdx.x & 31;
  const int half = lane >> 4;
  const int lm   = lane & 15;
  const int bx = blockIdx.x;
  const int qt = bx & 63;                  // S_LEN/32 = 64 query tiles
  const int h  = (bx >> 6) & (NH - 1);
  const int b  = bx >> 9;
  const int qbase = qt * 32;

  const __bf16* Q = qhb + (size_t)(b * NH + h) * S_LEN * HD;
  const __bf16* K = khb + (size_t)(b * NH + h) * S_LEN * HD;
  const __bf16* V = vTb + (size_t)(b * NH + h) * HD * S_LEN;

  // Q A-frags held in registers for the whole loop (lane: row lm, 16 d's)
  v16bf qa[2][2];
  #pragma unroll
  for (int qh = 0; qh < 2; ++qh) {
    qa[qh][0] = *(const v16bf*)(Q + (size_t)(qbase + qh * 16 + lm) * HD + half * 16);
    qa[qh][1] = *(const v16bf*)(Q + (size_t)(qbase + qh * 16 + lm) * HD + 32 + half * 16);
  }

  float mrun[2][8], lrun[2][8];
  v8f acc[2][4];
  #pragma unroll
  for (int qh = 0; qh < 2; ++qh) {
    #pragma unroll
    for (int r = 0; r < 8; ++r) { mrun[qh][r] = -3.0e38f; lrun[qh][r] = 0.0f; }
    #pragma unroll
    for (int nt = 0; nt < 4; ++nt) { v8f zz = {}; acc[qh][nt] = zz; }
  }

  for (int jb = 0; jb < S_LEN; jb += 32) {
    if (jb + 32 < S_LEN) {
      // next key block: 32 K rows (one cacheline each) via all 32 lanes,
      // 64 transposed-V rows via lanes {lane, lane+32}
      __builtin_prefetch(K + (size_t)(jb + 32 + lane) * HD, 0, 0);
      __builtin_prefetch(V + (size_t)lane * S_LEN + jb + 32, 0, 0);
      __builtin_prefetch(V + (size_t)(lane + 32) * S_LEN + jb + 32, 0, 0);
    }

    // K B-frags (column = key, 16 contiguous d's/lane) and V B-frags
    // (column = d, 16 contiguous keys/lane; V transposed [b,h,d,s]).
    // All issued up front: V latency hides behind score WMMAs + softmax.
    v16bf k0a = *(const v16bf*)(K + (size_t)(jb + lm) * HD + half * 16);
    v16bf k0b = *(const v16bf*)(K + (size_t)(jb + lm) * HD + 32 + half * 16);
    v16bf k1a = *(const v16bf*)(K + (size_t)(jb + 16 + lm) * HD + half * 16);
    v16bf k1b = *(const v16bf*)(K + (size_t)(jb + 16 + lm) * HD + 32 + half * 16);
    v16bf vb0 = *(const v16bf*)(V + (size_t)(0 * 16 + lm) * S_LEN + jb + half * 16);
    v16bf vb1 = *(const v16bf*)(V + (size_t)(1 * 16 + lm) * S_LEN + jb + half * 16);
    v16bf vb2 = *(const v16bf*)(V + (size_t)(2 * 16 + lm) * S_LEN + jb + half * 16);
    v16bf vb3 = *(const v16bf*)(V + (size_t)(3 * 16 + lm) * S_LEN + jb + half * 16);

    #pragma unroll
    for (int qh = 0; qh < 2; ++qh) {
      v8f s0 = {}; v8f s1 = {};
      s0 = wmma_bf16(qa[qh][0], k0a, s0);
      s0 = wmma_bf16(qa[qh][1], k0b, s0);
      s1 = wmma_bf16(qa[qh][0], k1a, s1);
      s1 = wmma_bf16(qa[qh][1], k1b, s1);

      const int j0 = jb + lm, j1 = jb + 16 + lm;
      float p0[8], p1[8], alpha[8];
      #pragma unroll
      for (int r = 0; r < 8; ++r) {
        const int i = qbase + qh * 16 + r + 8 * half;  // query row
        float v0 = s0[r] * 0.125f;                     // 1/sqrt(64)
        float v1 = s1[r] * 0.125f;
        int d0 = i - j0; d0 = d0 < 0 ? -d0 : d0;
        int d1 = i - j1; d1 = d1 < 0 ? -d1 : d1;
        if (d0 <= 5) v0 += 0.1f;                       // banded OCR bias
        if (d1 <= 5) v1 += 0.1f;
        // row max across the 16 lanes holding this row
        float t = fmaxf(v0, v1);
        t = fmaxf(t, __shfl_xor(t, 1));
        t = fmaxf(t, __shfl_xor(t, 2));
        t = fmaxf(t, __shfl_xor(t, 4));
        t = fmaxf(t, __shfl_xor(t, 8));
        float mn = fmaxf(mrun[qh][r], t);
        float a  = __expf(mrun[qh][r] - mn);
        float e0 = __expf(v0 - mn);
        float e1 = __expf(v1 - mn);
        float rs = e0 + e1;
        rs += __shfl_xor(rs, 1);
        rs += __shfl_xor(rs, 2);
        rs += __shfl_xor(rs, 4);
        rs += __shfl_xor(rs, 8);
        lrun[qh][r] = lrun[qh][r] * a + rs;
        mrun[qh][r] = mn;
        p0[r] = e0; p1[r] = e1; alpha[r] = a;
      }

      // rescale running output (row layout of acc matches score layout)
      #pragma unroll
      for (int nt = 0; nt < 4; ++nt)
        #pragma unroll
        for (int r = 0; r < 8; ++r)
          acc[qh][nt][r] *= alpha[r];

      // P: f32 D-layout -> bf16 A-layout via LDS (wave-private, dscnt only)
      #pragma unroll
      for (int r = 0; r < 8; ++r) {
        const int m = qh * 16 + r + 8 * half;
        pbuf[m * 32 + lm]      = (__bf16)p0[r];
        pbuf[m * 32 + 16 + lm] = (__bf16)p1[r];
      }
    }

    asm volatile("s_wait_dscnt 0" ::: "memory");
    v16bf pa0 = *(const v16bf*)(pbuf + (size_t)lm * 32 + half * 16);
    v16bf pa1 = *(const v16bf*)(pbuf + (size_t)(16 + lm) * 32 + half * 16);

    acc[0][0] = wmma_bf16(pa0, vb0, acc[0][0]);
    acc[1][0] = wmma_bf16(pa1, vb0, acc[1][0]);
    acc[0][1] = wmma_bf16(pa0, vb1, acc[0][1]);
    acc[1][1] = wmma_bf16(pa1, vb1, acc[1][1]);
    acc[0][2] = wmma_bf16(pa0, vb2, acc[0][2]);
    acc[1][2] = wmma_bf16(pa1, vb2, acc[1][2]);
    acc[0][3] = wmma_bf16(pa0, vb3, acc[0][3]);
    acc[1][3] = wmma_bf16(pa1, vb3, acc[1][3]);
  }

  // normalize and store context as bf16 [B, S, D] (K-contiguous for Wo GEMM)
  #pragma unroll
  for (int qh = 0; qh < 2; ++qh)
    #pragma unroll
    for (int nt = 0; nt < 4; ++nt)
      #pragma unroll
      for (int r = 0; r < 8; ++r) {
        const int m = qh * 16 + r + 8 * half;
        float o = acc[qh][nt][r] / lrun[qh][r];
        ctx[(size_t)(b * S_LEN + qbase + m) * DM + h * HD + nt * 16 + lm] = (__bf16)o;
      }
}

// ---------------------------------------------------------------------------
// Output projection: out = ctx @ Wo^T + bo, f32 result.  16x64 per wave.
// ---------------------------------------------------------------------------
__global__ void out_proj_kernel(const __bf16* __restrict__ ctx,
                                const __bf16* __restrict__ wob,
                                const float* __restrict__ bo,
                                float* __restrict__ out) {
  const int lane = threadIdx.x & 31;
  const int half = lane >> 4;
  const int lm   = lane & 15;
  const int colbase = blockIdx.x * 64;
  const int rowbase = blockIdx.y * 16;

  const __bf16* arow = ctx + (size_t)(rowbase + lm) * DM + half * 16;
  const __bf16* brow = wob + (size_t)(colbase + lm) * DM + half * 16;

  v8f acc0 = {}, acc1 = {}, acc2 = {}, acc3 = {};

  #pragma unroll 2
  for (int kk = 0; kk < DM; kk += 32) {
    v16bf a  = *(const v16bf*)(arow + kk);
    v16bf b0 = *(const v16bf*)(brow + (size_t)(0 * 16) * DM + kk);
    v16bf b1 = *(const v16bf*)(brow + (size_t)(1 * 16) * DM + kk);
    v16bf b2 = *(const v16bf*)(brow + (size_t)(2 * 16) * DM + kk);
    v16bf b3 = *(const v16bf*)(brow + (size_t)(3 * 16) * DM + kk);
    acc0 = wmma_bf16(a, b0, acc0);
    acc1 = wmma_bf16(a, b1, acc1);
    acc2 = wmma_bf16(a, b2, acc2);
    acc3 = wmma_bf16(a, b3, acc3);
    sched_loads_then_wmma();
  }

  v8f acc[4] = {acc0, acc1, acc2, acc3};
  #pragma unroll
  for (int nt = 0; nt < 4; ++nt) {
    const int c = colbase + nt * 16 + lm;
    const float bias = bo[c];
    #pragma unroll
    for (int r = 0; r < 8; ++r) {
      const int t = rowbase + r + 8 * half;
      out[(size_t)t * DM + c] = acc[nt][r] + bias;
    }
  }
}

// ---------------------------------------------------------------------------
// Launch
// ---------------------------------------------------------------------------
extern "C" void kernel_launch(void* const* d_in, const int* in_sizes, int n_in,
                              void* d_out, int out_size, void* d_ws, size_t ws_size,
                              hipStream_t stream) {
  (void)in_sizes; (void)n_in; (void)out_size; (void)ws_size;

  const float* x  = (const float*)d_in[0];
  // d_in[1] = ocr_context (unused by the reference computation)
  const float* Wq = (const float*)d_in[2];
  const float* bq = (const float*)d_in[3];
  const float* Wk = (const float*)d_in[4];
  const float* bk = (const float*)d_in[5];
  const float* Wv = (const float*)d_in[6];
  const float* bv = (const float*)d_in[7];
  const float* Wo = (const float*)d_in[8];
  const float* bo = (const float*)d_in[9];
  const float* ob = (const float*)d_in[10];
  float* out = (float*)d_out;

  const size_t nTok = (size_t)NTOK * DM;   // 8192*512 activation elems
  const size_t nW   = (size_t)DM * DM;

  char* p = (char*)d_ws;
  __bf16* xb  = (__bf16*)p; p += nTok * 2;
  __bf16* wqb = (__bf16*)p; p += nW * 2;
  __bf16* wkb = (__bf16*)p; p += nW * 2;
  __bf16* wvb = (__bf16*)p; p += nW * 2;
  __bf16* wob = (__bf16*)p; p += nW * 2;
  __bf16* qhb = (__bf16*)p; p += nTok * 2;   // [b,h,s,d]
  __bf16* khb = (__bf16*)p; p += nTok * 2;   // [b,h,s,d]
  __bf16* vTb = (__bf16*)p; p += nTok * 2;   // [b,h,d,s]
  __bf16* ctx = (__bf16*)p; p += nTok * 2;   // [b,s,D]

  // 1) downconvert activations + weights to bf16
  f32_to_bf16_kernel<<<(unsigned)((nTok + 255) / 256), 256, 0, stream>>>(x,  xb,  (int)nTok);
  f32_to_bf16_kernel<<<(unsigned)((nW   + 255) / 256), 256, 0, stream>>>(Wq, wqb, (int)nW);
  f32_to_bf16_kernel<<<(unsigned)((nW   + 255) / 256), 256, 0, stream>>>(Wk, wkb, (int)nW);
  f32_to_bf16_kernel<<<(unsigned)((nW   + 255) / 256), 256, 0, stream>>>(Wv, wvb, (int)nW);
  f32_to_bf16_kernel<<<(unsigned)((nW   + 255) / 256), 256, 0, stream>>>(Wo, wob, (int)nW);

  // 2) QKV projections (z selects Q/K/V), one wave per 16x64 tile
  dim3 gq(DM / 64, NTOK / 16, 3);
  qkv_proj_kernel<<<gq, 32, 0, stream>>>(xb, wqb, wkb, wvb, bq, bk, bv, ob,
                                         qhb, khb, vTb);

  // 3) flash attention: one wave per (b, h, 32-query tile)
  attn_fwd_kernel<<<BATCH * NH * (S_LEN / 32), 32, 0, stream>>>(qhb, khb, vTb, ctx);

  // 4) output projection -> f32
  out_proj_kernel<<<dim3(DM / 64, NTOK / 16), 32, 0, stream>>>(ctx, wob, bo, out);
}
